// F3SetMobileWrapper_81432579932478
// MI455X (gfx1250) — matrix operand order repaired and checked by reading, not compiled, not run on hardware
//
#include <hip/hip_runtime.h>
#include <hip/hip_bf16.h>

// ---------------- problem constants ----------------
#define B_SZ   256
#define T_LEN  96
#define F_DIM  368      // == D_MODEL (H)
#define H_DIM  368
#define KP     384      // K padded to multiple of 32 for 16x16x32 bf16 WMMA
#define NG     1104     // 3*H
#define NG_PAD 1152     // padded to 72 N-tiles (8 waves x 9 tiles)
#define NT_PW  9        // N-tiles per wave in the scan kernel
#define NCLS   29
#define M_ALL  (B_SZ * T_LEN)   // 24576

typedef __attribute__((ext_vector_type(8)))  float          v8f;
typedef __attribute__((ext_vector_type(16))) __bf16         v16bf;
typedef __attribute__((ext_vector_type(8)))  unsigned short ushort8;

union BfFrag {
    struct { ushort8 lo; ushort8 hi; } u;
    v16bf v;
};

__device__ __forceinline__ unsigned short f32_to_bf16(float f) {
    unsigned int u = __float_as_uint(f);
    unsigned int r = u + 0x7FFFu + ((u >> 16) & 1u);   // round-to-nearest-even
    return (unsigned short)(r >> 16);
}

__device__ __forceinline__ float sigmoid_f(float x) {
    return 1.0f / (1.0f + __expf(-x));
}

// ---------------- kernel 1: f32 -> bf16 with row+col zero padding ---------
// src: [Rsrc][C] f32 ; dst: [Rdst][Cp] bf16, zero beyond (Rsrc, C)
__global__ void cvt_pad_bf16(const float* __restrict__ src,
                             unsigned short* __restrict__ dst,
                             int Rsrc, int Rdst, int C, int Cp) {
    int i = blockIdx.x * blockDim.x + threadIdx.x;
    int total = Rdst * Cp;
    if (i >= total) return;
    int r = i / Cp, c = i - r * Cp;
    float v = (r < Rsrc && c < C) ? src[(size_t)r * C + c] : 0.0f;
    dst[i] = f32_to_bf16(v);
}

// ---------------- kernel 2: xproj = X @ Wi^T + bi  (bf16 WMMA, f32 acc) ----
// X  : [M][KP] bf16 (A, row-major)
// W  : [NG][KP] bf16 (B as N-major rows == Wi native layout, padded)
// out: [M][NG] f32
__global__ void xproj_gemm(const unsigned short* __restrict__ X,
                           const unsigned short* __restrict__ W,
                           const float* __restrict__ bi,
                           float* __restrict__ out) {
    const int nTiles = NG / 16;                 // 69
    int wave = threadIdx.x >> 5;
    int lane = threadIdx.x & 31;
    int tile = blockIdx.x * (blockDim.x >> 5) + wave;
    int mTile = tile / nTiles;
    int nTile = tile - mTile * nTiles;
    if (mTile * 16 >= M_ALL) return;            // wave-uniform

    int half = lane >> 4;
    int r    = lane & 15;

    const unsigned short* aRow = X + (size_t)(mTile * 16 + r) * KP + half * 8;
    const unsigned short* bRow = W + (size_t)(nTile * 16 + r) * KP + half * 8;

    v8f acc = {0.f, 0.f, 0.f, 0.f, 0.f, 0.f, 0.f, 0.f};
    for (int k0 = 0; k0 < KP; k0 += 32) {
        BfFrag a, b;
        a.u.lo = *(const ushort8*)(aRow + k0);
        a.u.hi = *(const ushort8*)(aRow + k0 + 16);
        b.u.lo = *(const ushort8*)(bRow + k0);
        b.u.hi = *(const ushort8*)(bRow + k0 + 16);
        acc = __builtin_amdgcn_wmma_f32_16x16x32_bf16(
            false, a.v, false, b.v, (short)0, acc, false, false);
    }

    int n = nTile * 16 + r;
    float bias = bi[n];
    int mBase = mTile * 16 + half * 8;
#pragma unroll
    for (int i = 0; i < 8; ++i)
        out[(size_t)(mBase + i) * NG + n] = acc[i] + bias;
}

// ---------------- kernel 3: GRU scan ----------------
// One block owns 16 batch rows for all T steps. h kept in LDS (f32 + bf16
// copy as WMMA A operand). Wh (bf16, [NG_PAD][KP]) streamed from L2 each
// step. k0-outer / nTile-inner schedule: one LDS A-fragment load feeds 9
// back-to-back WMMAs (9 live v8f accumulators per wave).
__global__ void gru_scan(const unsigned short* __restrict__ WhB,
                         const float* __restrict__ Xproj,   // [M][NG]
                         const float* __restrict__ bh,      // [NG]
                         float* __restrict__ temporal) {    // [M][H]
    __shared__ __align__(16) unsigned short h_bf[16 * KP];  // 12 KB
    __shared__ float gh[16 * NG];                           // 70.6 KB
    __shared__ float hf[16 * H_DIM];                        // 23.5 KB

    const int tid  = threadIdx.x;
    const int wave = tid >> 5;
    const int lane = tid & 31;
    const int half = lane >> 4;
    const int r    = lane & 15;
    const int b0   = blockIdx.x * 16;

    for (int i = tid; i < 16 * KP; i += 256) h_bf[i] = 0;
    for (int i = tid; i < 16 * H_DIM; i += 256) hf[i] = 0.0f;
    __syncthreads();

    const unsigned short* aRow = h_bf + r * KP + half * 8;
    // this wave's 9 consecutive N-tiles: [wave*9, wave*9+9)
    const unsigned short* bBase =
        WhB + (size_t)(wave * NT_PW * 16 + r) * KP + half * 8;

    for (int t = 0; t < T_LEN; ++t) {
        // ---- gh = h @ Wh^T + bh via WMMA ----
        v8f acc[NT_PW];
#pragma unroll
        for (int j = 0; j < NT_PW; ++j)
            acc[j] = (v8f){0.f, 0.f, 0.f, 0.f, 0.f, 0.f, 0.f, 0.f};

        for (int k0 = 0; k0 < KP; k0 += 32) {
            BfFrag a;
            a.u.lo = *(const ushort8*)(aRow + k0);
            a.u.hi = *(const ushort8*)(aRow + k0 + 16);
#pragma unroll
            for (int j = 0; j < NT_PW; ++j) {
                const unsigned short* bRow = bBase + (size_t)j * 16 * KP + k0;
                BfFrag b;
                b.u.lo = *(const ushort8*)(bRow);
                b.u.hi = *(const ushort8*)(bRow + 16);
                acc[j] = __builtin_amdgcn_wmma_f32_16x16x32_bf16(
                    false, a.v, false, b.v, (short)0, acc[j], false, false);
            }
        }

#pragma unroll
        for (int j = 0; j < NT_PW; ++j) {
            int n = (wave * NT_PW + j) * 16 + r;
            if (n < NG) {                       // wave-uniform per tile
                float bias = bh[n];
#pragma unroll
                for (int i = 0; i < 8; ++i)
                    gh[(half * 8 + i) * NG + n] = acc[j][i] + bias;
            }
        }
        __syncthreads();

        // ---- gate math: 16 x 368 elements across 256 threads ----
        for (int e = tid; e < 16 * H_DIM; e += 256) {
            int row = e / H_DIM, col = e - row * H_DIM;
            size_t gm = (size_t)(b0 + row) * T_LEN + t;
            const float* xp = Xproj + gm * NG;
            float gi_r = xp[col];
            float gi_z = xp[H_DIM + col];
            float gi_n = xp[2 * H_DIM + col];
            float rr = sigmoid_f(gi_r + gh[row * NG + col]);
            float zz = sigmoid_f(gi_z + gh[row * NG + H_DIM + col]);
            float nn = tanhf(gi_n + rr * gh[row * NG + 2 * H_DIM + col]);
            float hn = (1.0f - zz) * nn + zz * hf[e];
            hf[e] = hn;
            h_bf[row * KP + col] = f32_to_bf16(hn);
            temporal[gm * H_DIM + col] = hn;
        }
        __syncthreads();
    }
}

// ---------------- kernel 4: output heads ----------------
__global__ void heads_kernel(const float* __restrict__ temporal,
                             const float* __restrict__ Wc, const float* __restrict__ bc,
                             const float* __restrict__ Wf, const float* __restrict__ bfv,
                             float* __restrict__ cs,     // [M][2] softmaxed
                             float* __restrict__ fine) { // [M][29] sigmoided
    int m = blockIdx.x * blockDim.x + threadIdx.x;
    if (m >= M_ALL) return;
    const float* h = temporal + (size_t)m * H_DIM;
    float c0 = bc[0], c1 = bc[1];
    float accF[NCLS];
#pragma unroll
    for (int c = 0; c < NCLS; ++c) accF[c] = bfv[c];
    for (int k = 0; k < H_DIM; ++k) {
        float x = h[k];
        c0 += x * Wc[k];
        c1 += x * Wc[H_DIM + k];
#pragma unroll
        for (int c = 0; c < NCLS; ++c) accF[c] += x * Wf[c * H_DIM + k];
    }
    float mx = fmaxf(c0, c1);
    float e0 = __expf(c0 - mx), e1 = __expf(c1 - mx);
    float inv = 1.0f / (e0 + e1);
    cs[(size_t)m * 2]     = e0 * inv;
    cs[(size_t)m * 2 + 1] = e1 * inv;
#pragma unroll
    for (int c = 0; c < NCLS; ++c)
        fine[(size_t)m * NCLS + c] = sigmoid_f(accF[c]);
}

// ---------------- kernel 5: NMS + argmax ----------------
__global__ void nms_kernel(const float* __restrict__ cs,
                           float* __restrict__ dec,       // [M]
                           float* __restrict__ out_nms) { // [M][2]
    int idx = blockIdx.x * blockDim.x + threadIdx.x;
    if (idx >= M_ALL) return;
    int b = idx / T_LEN, t = idx - b * T_LEN;
    float bg = cs[(size_t)idx * 2];
    float wmin = bg;
#pragma unroll
    for (int d = -2; d <= 2; ++d) {
        int tt = t + d;
        if (tt >= 0 && tt < T_LEN)
            wmin = fminf(wmin, cs[((size_t)b * T_LEN + tt) * 2]);
    }
    bool keep = (bg == wmin);
    float n0 = keep ? bg : 0.0f;
    float n1 = keep ? cs[(size_t)idx * 2 + 1] : 0.0f;
    out_nms[(size_t)idx * 2]     = n0;
    out_nms[(size_t)idx * 2 + 1] = n1;
    dec[idx] = (n1 > n0) ? 1.0f : 0.0f;   // jnp.argmax: first index on ties
}

// ---------------- host launcher ----------------
extern "C" void kernel_launch(void* const* d_in, const int* in_sizes, int n_in,
                              void* d_out, int out_size, void* d_ws, size_t ws_size,
                              hipStream_t stream) {
    const float* features = (const float*)d_in[0];  // (B,T,F)
    // d_in[1] = hand (unused, matching reference)
    const float* gru_Wi = (const float*)d_in[2];    // (3H, F)
    const float* gru_Wh = (const float*)d_in[3];    // (3H, H)
    const float* gru_bi = (const float*)d_in[4];    // (3H)
    const float* gru_bh = (const float*)d_in[5];    // (3H)
    const float* Wc = (const float*)d_in[6];        // (2, H)
    const float* bc = (const float*)d_in[7];        // (2)
    const float* Wf = (const float*)d_in[8];        // (29, H)
    const float* bfv = (const float*)d_in[9];       // (29)

    // workspace carve-up (256B aligned)
    char* ws = (char*)d_ws;
    size_t off = 0;
    auto carve = [&](size_t bytes) {
        char* p = ws + off;
        off = (off + bytes + 255) & ~(size_t)255;
        return p;
    };
    unsigned short* X_bf  = (unsigned short*)carve((size_t)M_ALL * KP * 2);
    unsigned short* Wi_bf = (unsigned short*)carve((size_t)NG * KP * 2);
    unsigned short* Wh_bf = (unsigned short*)carve((size_t)NG_PAD * KP * 2);
    float* Xproj    = (float*)carve((size_t)M_ALL * NG * 4);
    float* temporal = (float*)carve((size_t)M_ALL * H_DIM * 4);
    float* cs       = (float*)carve((size_t)M_ALL * 2 * 4);

    float* out = (float*)d_out;
    float* out_dec  = out;                 // [M]
    float* out_nms  = out + M_ALL;         // [M][2]
    float* out_fine = out + M_ALL * 3;     // [M][29]

    // 1) convert + pad to bf16
    {
        int total = M_ALL * KP;
        cvt_pad_bf16<<<(total + 255) / 256, 256, 0, stream>>>(
            features, X_bf, M_ALL, M_ALL, F_DIM, KP);
    }
    {
        int totalWi = NG * KP;
        cvt_pad_bf16<<<(totalWi + 255) / 256, 256, 0, stream>>>(
            gru_Wi, Wi_bf, NG, NG, F_DIM, KP);
        int totalWh = NG_PAD * KP;
        cvt_pad_bf16<<<(totalWh + 255) / 256, 256, 0, stream>>>(
            gru_Wh, Wh_bf, NG, NG_PAD, H_DIM, KP);
    }

    // 2) xproj GEMM: (M_ALL/16)*(NG/16) = 1536*69 = 105984 wave-tiles, 8/block
    {
        int tiles = (M_ALL / 16) * (NG / 16);
        xproj_gemm<<<tiles / 8, 256, 0, stream>>>(X_bf, Wi_bf, gru_bi, Xproj);
    }

    // 3) sequential GRU scan: 16 blocks, each owns 16 batch rows
    gru_scan<<<B_SZ / 16, 256, 0, stream>>>(Wh_bf, Xproj, gru_bh, temporal);

    // 4) heads
    heads_kernel<<<(M_ALL + 255) / 256, 256, 0, stream>>>(
        temporal, Wc, bc, Wf, bfv, cs, out_fine);

    // 5) NMS + argmax
    nms_kernel<<<(M_ALL + 255) / 256, 256, 0, stream>>>(cs, out_dec, out_nms);
}
